// ReformerEncoder_8083128451369
// MI455X (gfx1250) — matrix operand order repaired and checked by reading, not compiled, not run on hardware
//
#include <hip/hip_runtime.h>

// ---------------------------------------------------------------------------
// Reformer encoder forward for gfx1250 (MI455X).
// Dense GEMMs: v_wmma_f32_16x16x32_bf16 with async global->LDS B-tile staging
// (double buffered, ASYNCcnt). LSH bucketing / sort / softmax in f32.
// ---------------------------------------------------------------------------

#define DEPTH   2
#define HEADS   8
#define DIMM    1024
#define DHEAD   128
#define SEQT    4096
#define BUCKETS 64      // NB (and bucket count per hash)
#define NHASH   4
#define BATCH   2
#define FFDIM   4096
#define BT      (BATCH * SEQT)        // 8192 rows
#define BHN     (BATCH * HEADS)       // 16
#define NHT     (NHASH * SEQT)        // 16384
#define NCHUNK  (NHASH * BUCKETS)     // 256 chunks of 64

typedef __attribute__((ext_vector_type(16))) __bf16 v16bf;
typedef __attribute__((ext_vector_type(8)))  __bf16 v8bf;
typedef __attribute__((ext_vector_type(8)))  float  v8f;

// ---------------------------------------------------------------------------
// helpers
// ---------------------------------------------------------------------------
__device__ __forceinline__ float gelu_f(float x) {
    const float c = 0.7978845608028654f; // sqrt(2/pi)
    float x3 = x * x * x;
    return 0.5f * x * (1.0f + tanhf(c * (x + 0.044715f * x3)));
}

// Build a 16x32 bf16 A-fragment: lane holds K[koff..koff+16),
// koff = (lane>>4)*16  (contiguous k order, matches B).
__device__ __forceinline__ v16bf load_a_frag(const __bf16* rowptr, int koffA) {
    v8bf alo = *(const v8bf*)(rowptr + koffA);
    v8bf ahi = *(const v8bf*)(rowptr + koffA + 8);
    v16bf a;
#pragma unroll
    for (int i = 0; i < 8; ++i) { a[i] = alo[i]; a[i + 8] = ahi[i]; }
    return a;
}

// Async DMA one 64x32 bf16 B-tile slice (4KB) into LDS: one b128 per thread.
// GVS addressing: saddr = Bt base, voffset = byte offset (<= 32MB here).
__device__ __forceinline__ void async_stage_b(unsigned long long bt_base, int K,
                                              int n0, int k0, __bf16* dst, int tid)
{
    int row = tid >> 2;            // 0..63  (B column index within tile)
    int chunk = tid & 3;           // 4 x 8 bf16 chunks per 32-wide k slice
    unsigned voff = (unsigned)((((size_t)(n0 + row) * K) + k0 + chunk * 8) * 2);
    unsigned lds  = (unsigned)(uintptr_t)(dst + row * 32 + chunk * 8);
    asm volatile("global_load_async_to_lds_b128 %0, %1, %2 offset:0"
                 :: "v"(lds), "v"(voff), "s"(bt_base) : "memory");
}

__device__ __forceinline__ void wait_async_zero() {
    asm volatile("s_wait_asynccnt 0x0" ::: "memory");
}

// ---------------------------------------------------------------------------
// Weight transpose + cast: src[K,N] f32 -> dst[N,K] bf16  (grid = K*N/256)
// ---------------------------------------------------------------------------
__global__ __launch_bounds__(256) void cast_transpose_kernel(
    const float* __restrict__ src, __bf16* __restrict__ dst, int K, int N)
{
    size_t i = (size_t)blockIdx.x * 256 + threadIdx.x;
    int k = (int)(i / N);
    int n = (int)(i % N);
    dst[(size_t)n * K + k] = (__bf16)src[i];
}

// ---------------------------------------------------------------------------
// init / final
// ---------------------------------------------------------------------------
__global__ __launch_bounds__(256) void init_streams_kernel(
    const float* __restrict__ x, float* __restrict__ x1, float* __restrict__ x2)
{
    size_t i = (size_t)blockIdx.x * 256 + threadIdx.x;
    float v = x[i];
    x1[i] = v; x2[i] = v;
}

__global__ __launch_bounds__(256) void final_kernel(
    const float* __restrict__ x1, const float* __restrict__ x2, float* __restrict__ out)
{
    size_t i = (size_t)blockIdx.x * 256 + threadIdx.x;
    out[i] = (x1[i] + x2[i]) * 0.5f;
}

// ---------------------------------------------------------------------------
// LayerNorm over DIM=1024, one row per block, bf16 output for WMMA GEMMs
// ---------------------------------------------------------------------------
__global__ __launch_bounds__(256) void layernorm_kernel(
    const float* __restrict__ x, const float* __restrict__ g,
    const float* __restrict__ b, __bf16* __restrict__ out)
{
    const int row = blockIdx.x;
    const float* xr = x + (size_t)row * DIMM;
    __shared__ float red[256];
    float s = 0.f;
    for (int i = threadIdx.x; i < DIMM; i += 256) s += xr[i];
    red[threadIdx.x] = s; __syncthreads();
    for (int st = 128; st > 0; st >>= 1) {
        if (threadIdx.x < st) red[threadIdx.x] += red[threadIdx.x + st];
        __syncthreads();
    }
    float mu = red[0] * (1.0f / DIMM);
    __syncthreads();
    float vv = 0.f;
    for (int i = threadIdx.x; i < DIMM; i += 256) { float d = xr[i] - mu; vv += d * d; }
    red[threadIdx.x] = vv; __syncthreads();
    for (int st = 128; st > 0; st >>= 1) {
        if (threadIdx.x < st) red[threadIdx.x] += red[threadIdx.x + st];
        __syncthreads();
    }
    float rstd = rsqrtf(red[0] * (1.0f / DIMM) + 1e-5f);
    for (int i = threadIdx.x; i < DIMM; i += 256)
        out[(size_t)row * DIMM + i] = (__bf16)((xr[i] - mu) * rstd * g[i] + b[i]);
}

// ---------------------------------------------------------------------------
// Generic bf16 WMMA GEMM: C[M,N] = A[M,K] @ Bt[N,K]^T (+epilogue)
//   EPI 0: f32 store (HSPLIT => scatter into [bh,T,DH] head layout)
//   EPI 1: f32 store of acc + bias[n] + resid[m,n]
//   EPI 2: bf16 store of gelu(acc + bias[n])
// block = 256 thr (8 waves); tile 128(M) x 64(N); wave = 16 x 64; k-step 32.
// B k-slices are DMA'd into LDS with global_load_async_to_lds_b128 (double
// buffered; prefetch of stage k+1 overlaps WMMA of stage k).
// ---------------------------------------------------------------------------
template<int EPI, bool HSPLIT>
__global__ __launch_bounds__(256) void gemm_bf16_kernel(
    const __bf16* __restrict__ A, const __bf16* __restrict__ Bt,
    float* __restrict__ Cf, __bf16* __restrict__ Cb,
    const float* __restrict__ bias, const float* __restrict__ resid,
    int M, int N, int K)
{
    const int tid  = threadIdx.x;
    const int lane = tid & 31;
    const int wave = tid >> 5;
    const int m0 = blockIdx.y * 128 + wave * 16;
    const int n0 = blockIdx.x * 64;
    const int rowA  = m0 + (lane & 15);
    const int koffA = (lane >> 4) * 16;
    const int kbB   = (lane >> 4) * 16;

    __shared__ __bf16 bsh[2][64 * 32];   // double-buffered B tile (2 x 4KB)

    v8f acc[4];
#pragma unroll
    for (int j = 0; j < 4; ++j)
#pragma unroll
        for (int r = 0; r < 8; ++r) acc[j][r] = 0.f;

    const unsigned long long bt_base = (unsigned long long)(uintptr_t)Bt;
    const int nk = K >> 5;

    async_stage_b(bt_base, K, n0, 0, &bsh[0][0], tid);

    for (int kt = 0; kt < nk; ++kt) {
        wait_async_zero();      // our wave's stage-kt DMAs landed
        __syncthreads();        // all waves' stage-kt DMAs visible
        if (kt + 1 < nk)
            async_stage_b(bt_base, K, n0, (kt + 1) << 5, &bsh[(kt + 1) & 1][0], tid);

        const __bf16* bbuf = &bsh[kt & 1][0];
        const int k0 = kt << 5;
        v16bf a = load_a_frag(A + (size_t)rowA * K + k0, koffA);
#pragma unroll
        for (int j = 0; j < 4; ++j) {
            int col = j * 16 + (lane & 15);
            v16bf b = *(const v16bf*)(bbuf + col * 32 + kbB);
            acc[j] = __builtin_amdgcn_wmma_f32_16x16x32_bf16(
                false, a, false, b, (short)0, acc[j], false, false);
        }
    }

    const int mbase = m0 + ((lane >> 4) << 3);
#pragma unroll
    for (int j = 0; j < 4; ++j) {
        int n = n0 + j * 16 + (lane & 15);
#pragma unroll
        for (int r = 0; r < 8; ++r) {
            int m = mbase + r;
            float val = acc[j][r];
            if (EPI == 1) val += bias[n] + resid[(size_t)m * N + n];
            if (EPI == 2) {
                val = gelu_f(val + bias[n]);
                Cb[(size_t)m * N + n] = (__bf16)val;
            } else if (HSPLIT) {
                int bb = m >> 12, tt = m & (SEQT - 1);
                int hh = n >> 7, dd = n & (DHEAD - 1);
                Cf[(((size_t)bb * HEADS + hh) * SEQT + tt) * DHEAD + dd] = val;
            } else {
                Cf[(size_t)m * N + n] = val;
            }
        }
    }
}

// ---------------------------------------------------------------------------
// LSH rotation + argmax(rot, -rot) -> bucket id per (bh, hash, t). f32 exact.
// grid (SEQT, BHN), 128 threads.
// ---------------------------------------------------------------------------
__global__ __launch_bounds__(128) void rotate_argmax_kernel(
    const float* __restrict__ qk, const float* __restrict__ rot,
    unsigned char* __restrict__ buckets)
{
    const int t = blockIdx.x, bh = blockIdx.y;
    __shared__ float qs[DHEAD];
    __shared__ float rr[128];
    qs[threadIdx.x] = qk[((size_t)bh * SEQT + t) * DHEAD + threadIdx.x];
    __syncthreads();
    float s = 0.f;
#pragma unroll 4
    for (int k = 0; k < DHEAD; ++k) s += qs[k] * rot[k * 128 + threadIdx.x];
    rr[threadIdx.x] = s;
    __syncthreads();
    if (threadIdx.x < NHASH) {
        int h = threadIdx.x;
        float best = -3.4e38f; int bi = 0;
        for (int j = 0; j < BUCKETS; ++j) {
            float v = (j < 32) ? rr[h * 32 + j] : -rr[h * 32 + j - 32];
            if (v > best) { best = v; bi = j; }
        }
        buckets[(size_t)bh * NHT + (size_t)h * SEQT + t] = (unsigned char)bi;
    }
}

// ---------------------------------------------------------------------------
// Stable counting sort by bucket (ties broken by t) per (bh, hash).
// st[p] = original t; undo[t] = global sorted position (h*T + p).
// grid (NHASH, BHN), 256 threads.
// ---------------------------------------------------------------------------
__global__ __launch_bounds__(256) void bucket_sort_kernel(
    const unsigned char* __restrict__ buckets, int* __restrict__ st, int* __restrict__ undo)
{
    const int h = blockIdx.x, bh = blockIdx.y;
    __shared__ int cnt[BUCKETS], offs[BUCKETS];
    __shared__ unsigned char bks[SEQT];
    if (threadIdx.x < BUCKETS) cnt[threadIdx.x] = 0;
    __syncthreads();
    const unsigned char* bsrc = buckets + (size_t)bh * NHT + (size_t)h * SEQT;
    for (int t = threadIdx.x; t < SEQT; t += 256) {
        unsigned char bb = bsrc[t];
        bks[t] = bb;
        atomicAdd(&cnt[bb], 1);
    }
    __syncthreads();
    if (threadIdx.x == 0) {
        int run = 0;
        for (int i = 0; i < BUCKETS; ++i) { offs[i] = run; run += cnt[i]; }
    }
    __syncthreads();
    int* stp = st   + (size_t)bh * NHT + (size_t)h * SEQT;
    int* unp = undo + (size_t)bh * NHT + (size_t)h * SEQT;
    for (int s0 = 0; s0 < SEQT; s0 += 256) {
        int t  = s0 + threadIdx.x;
        int bb = bks[t];
        int rank = 0;
        for (int j = 0; j < threadIdx.x; ++j) rank += (bks[s0 + j] == bb) ? 1 : 0;
        int pos = offs[bb] + rank;
        stp[pos] = t;
        unp[t]   = h * SEQT + pos;
        __syncthreads();
        atomicAdd(&offs[bb], 1);
        __syncthreads();
    }
}

// ---------------------------------------------------------------------------
// Chunked LSH attention: one block per (chunk c, bh). 64 queries attend to
// 128 keys (chunk c + chunk c-1 mod NCHUNK). WMMA for qk^T and probs@v.
// ---------------------------------------------------------------------------
__global__ __launch_bounds__(256) void attn_chunk_kernel(
    const float* __restrict__ qk, const float* __restrict__ v,
    const int* __restrict__ st, __bf16* __restrict__ bo_sorted,
    float* __restrict__ lse_sorted)
{
    const int c   = blockIdx.x;
    const int bh  = blockIdx.y;
    const int tid = threadIdx.x;

    __shared__ __bf16 ab_s[64][128];   // bq, then probs (e-values)
    __shared__ __bf16 kv_s[128][128];  // bk[key][d], then bvT[d][key]
    __shared__ float  rinv_s[128];
    __shared__ int    bt_s[64];
    __shared__ int    bkt_s[128];
    __shared__ float  redmax[2][64];
    __shared__ float  redsum[2][64];
    __shared__ float  rsuminv[64];

    const int* stp = st + (size_t)bh * NHT;
    for (int j = tid; j < 128; j += 256) {
        int cc = (j < 64) ? c : ((c + NCHUNK - 1) & (NCHUNK - 1));
        int tk = stp[cc * 64 + (j & 63)];
        bkt_s[j] = tk;
        if (j < 64) bt_s[j] = tk;
    }
    __syncthreads();

    const float* qkb = qk + (size_t)bh * SEQT * DHEAD;
    for (int j = tid; j < 128; j += 256) {
        const float* row = qkb + (size_t)bkt_s[j] * DHEAD;
        float ss = 1e-12f;
#pragma unroll 4
        for (int d2 = 0; d2 < DHEAD; ++d2) { float q = row[d2]; ss += q * q; }
        rinv_s[j] = rsqrtf(ss);
    }
    __syncthreads();
    for (int i = tid; i < 128 * 128; i += 256) {
        int j = i >> 7, d2 = i & 127;
        float q = qkb[(size_t)bkt_s[j] * DHEAD + d2];
        kv_s[j][d2] = (__bf16)(q * rinv_s[j]);
        if (j < 64) ab_s[j][d2] = (__bf16)q;
    }
    __syncthreads();

    const int lane = tid & 31, wave = tid >> 5;
    const int mt = wave & 3;          // m-tile 0..3
    const int ng = wave >> 2;         // n-half 0..1 (64 cols each)
    const int arow  = mt * 16 + (lane & 15);
    const int koffA = (lane >> 4) * 16;
    const int kbB   = (lane >> 4) * 16;
    const int mbase = mt * 16 + ((lane >> 4) << 3);

    // ---- dots = bq @ bk^T ----
    v8f acc[4];
#pragma unroll
    for (int j = 0; j < 4; ++j)
#pragma unroll
        for (int r = 0; r < 8; ++r) acc[j][r] = 0.f;

#pragma unroll
    for (int k0 = 0; k0 < 128; k0 += 32) {
        v16bf a = load_a_frag(&ab_s[arow][k0], koffA);
#pragma unroll
        for (int j = 0; j < 4; ++j) {
            int coln = ng * 64 + j * 16 + (lane & 15);
            v16bf b = *(const v16bf*)(&kv_s[coln][k0 + kbB]);
            acc[j] = __builtin_amdgcn_wmma_f32_16x16x32_bf16(
                false, a, false, b, (short)0, acc[j], false, false);
        }
    }

    // scale + self-mask
    const float scale = 0.08838834764831845f; // 128^-0.5
#pragma unroll
    for (int j = 0; j < 4; ++j) {
        int n = ng * 64 + j * 16 + (lane & 15);
        int tkn = bkt_s[n];
#pragma unroll
        for (int r = 0; r < 8; ++r) {
            float val = acc[j][r] * scale;
            if (bt_s[mbase + r] == tkn) val -= 1e5f;
            acc[j][r] = val;
        }
    }

    // partial row max across this wave's 64 columns
    float rmax[8];
#pragma unroll
    for (int r = 0; r < 8; ++r) {
        float mv = fmaxf(fmaxf(acc[0][r], acc[1][r]), fmaxf(acc[2][r], acc[3][r]));
#pragma unroll
        for (int msk = 1; msk < 16; msk <<= 1) mv = fmaxf(mv, __shfl_xor(mv, msk, 32));
        rmax[r] = mv;
    }
    if ((lane & 15) == 0) {
#pragma unroll
        for (int r = 0; r < 8; ++r) redmax[ng][mbase + r] = rmax[r];
    }
    __syncthreads();
    if (tid < 64) redmax[0][tid] = fmaxf(redmax[0][tid], redmax[1][tid]);
    __syncthreads();

    // exp, partial sums, write e-values back into ab_s (probs * rowsum)
    float rsum[8];
#pragma unroll
    for (int r = 0; r < 8; ++r) {
        float gm = redmax[0][mbase + r];
        float s = 0.f;
#pragma unroll
        for (int j = 0; j < 4; ++j) {
            int n = ng * 64 + j * 16 + (lane & 15);
            float e = expf(acc[j][r] - gm);
            ab_s[mbase + r][n] = (__bf16)e;
            s += e;
        }
#pragma unroll
        for (int msk = 1; msk < 16; msk <<= 1) s += __shfl_xor(s, msk, 32);
        rsum[r] = s;
    }
    if ((lane & 15) == 0) {
#pragma unroll
        for (int r = 0; r < 8; ++r) redsum[ng][mbase + r] = rsum[r];
    }
    __syncthreads();
    if (tid < 64) {
        float tot = redsum[0][tid] + redsum[1][tid];
        rsuminv[tid] = 1.0f / tot;
        lse_sorted[(size_t)bh * NHT + (size_t)c * 64 + tid] = redmax[0][tid] + logf(tot);
    }
    __syncthreads();

    // reload kv_s with bv2 transposed: kv_s[d][key]
    const float* vb = v + (size_t)bh * SEQT * DHEAD;
    for (int i = tid; i < 128 * 128; i += 256) {
        int j = i >> 7, d2 = i & 127;
        kv_s[d2][j] = (__bf16)vb[(size_t)bkt_s[j] * DHEAD + d2];
    }
    __syncthreads();

    // ---- bo = probs @ bv2 ----
    v8f acc2[4];
#pragma unroll
    for (int j = 0; j < 4; ++j)
#pragma unroll
        for (int r = 0; r < 8; ++r) acc2[j][r] = 0.f;

#pragma unroll
    for (int k0 = 0; k0 < 128; k0 += 32) {
        v16bf a = load_a_frag(&ab_s[arow][k0], koffA);
#pragma unroll
        for (int j = 0; j < 4; ++j) {
            int coln = ng * 64 + j * 16 + (lane & 15);
            v16bf b = *(const v16bf*)(&kv_s[coln][k0 + kbB]);
            acc2[j] = __builtin_amdgcn_wmma_f32_16x16x32_bf16(
                false, a, false, b, (short)0, acc2[j], false, false);
        }
    }

    __bf16* bop = bo_sorted + ((size_t)bh * NHT + (size_t)c * 64) * DHEAD;
#pragma unroll
    for (int j = 0; j < 4; ++j) {
        int n = ng * 64 + j * 16 + (lane & 15);
#pragma unroll
        for (int r = 0; r < 8; ++r) {
            int m = mbase + r;
            bop[(size_t)m * DHEAD + n] = (__bf16)(acc2[j][r] * rsuminv[m]);
        }
    }
}

// ---------------------------------------------------------------------------
// Unsort + combine hash rounds with softmax(lse) weights -> attn bf16
// grid (SEQT, BHN), 128 threads (one per d).
// ---------------------------------------------------------------------------
__global__ __launch_bounds__(128) void combine_kernel(
    const __bf16* __restrict__ bo_sorted, const float* __restrict__ lse_sorted,
    const int* __restrict__ undo, __bf16* __restrict__ attn_bf)
{
    const int t = blockIdx.x, bh = blockIdx.y;
    const int d = threadIdx.x;
    __shared__ float lg[NHASH];
    __shared__ int   pos_s[NHASH];
    const int* unp = undo + (size_t)bh * NHT;
    if (d < NHASH) {
        int p = unp[d * SEQT + t];
        pos_s[d] = p;
        lg[d] = lse_sorted[(size_t)bh * NHT + p];
    }
    __syncthreads();
    float mx = fmaxf(fmaxf(lg[0], lg[1]), fmaxf(lg[2], lg[3]));
    float e0 = expf(lg[0] - mx), e1 = expf(lg[1] - mx);
    float e2 = expf(lg[2] - mx), e3 = expf(lg[3] - mx);
    float inv = 1.0f / (e0 + e1 + e2 + e3);
    float w[NHASH] = { e0 * inv, e1 * inv, e2 * inv, e3 * inv };
    float o = 0.f;
#pragma unroll
    for (int h = 0; h < NHASH; ++h)
        o += (float)bo_sorted[((size_t)bh * NHT + pos_s[h]) * DHEAD + d] * w[h];
    int b = bh >> 3, head = bh & 7;
    attn_bf[((size_t)b * SEQT + t) * DIMM + head * DHEAD + d] = (__bf16)o;
}

// ---------------------------------------------------------------------------
// host orchestration
// ---------------------------------------------------------------------------
extern "C" void kernel_launch(void* const* d_in, const int* in_sizes, int n_in,
                              void* d_out, int out_size, void* d_ws, size_t ws_size,
                              hipStream_t stream) {
    (void)in_sizes; (void)n_in; (void)out_size; (void)ws_size;
    const float* x    = (const float*)d_in[0];
    // d_in[1] = mask (all ones; unused by the reference attention path)
    const float* ln1g = (const float*)d_in[2];
    const float* ln1b = (const float*)d_in[3];
    const float* Wqk  = (const float*)d_in[4];
    const float* Wv   = (const float*)d_in[5];
    const float* Wo   = (const float*)d_in[6];
    const float* bo   = (const float*)d_in[7];
    const float* ln2g = (const float*)d_in[8];
    const float* ln2b = (const float*)d_in[9];
    const float* W1   = (const float*)d_in[10];
    const float* b1   = (const float*)d_in[11];
    const float* W2   = (const float*)d_in[12];
    const float* b2   = (const float*)d_in[13];
    const float* rot  = (const float*)d_in[14];

    char* ws = (char*)d_ws;
    size_t off = 0;
    auto take = [&](size_t bytes) -> void* {
        void* p = ws + off;
        off += (bytes + 255) & ~(size_t)255;
        return p;
    };

    __bf16* wqkB = (__bf16*)take((size_t)DEPTH * DIMM * DIMM * 2);
    __bf16* wvB  = (__bf16*)take((size_t)DEPTH * DIMM * DIMM * 2);
    __bf16* woB  = (__bf16*)take((size_t)DEPTH * DIMM * DIMM * 2);
    __bf16* w1B  = (__bf16*)take((size_t)DEPTH * DIMM * FFDIM * 2);
    __bf16* w2B  = (__bf16*)take((size_t)DEPTH * FFDIM * DIMM * 2);
    float*  x1   = (float*)take((size_t)BT * DIMM * 4);
    float*  x2   = (float*)take((size_t)BT * DIMM * 4);
    __bf16* hbf  = (__bf16*)take((size_t)BT * DIMM * 2);       // LN out / attn out
    float*  qkb  = (float*)take((size_t)BHN * SEQT * DHEAD * 4);
    float*  vb   = (float*)take((size_t)BHN * SEQT * DHEAD * 4);
    unsigned char* bkb = (unsigned char*)take((size_t)BHN * NHT);
    int*    stb  = (int*)take((size_t)BHN * NHT * 4);
    int*    unb  = (int*)take((size_t)BHN * NHT * 4);
    float*  lseb = (float*)take((size_t)BHN * NHT * 4);
    __bf16* boff = (__bf16*)take((size_t)BHN * NHT * DHEAD * 2); // bo_sorted / ff1

    // weight transpose+cast (once per launch; deterministic)
    for (int l = 0; l < DEPTH; ++l) {
        size_t o1 = (size_t)l * DIMM * DIMM;
        size_t o2 = (size_t)l * DIMM * FFDIM;
        cast_transpose_kernel<<<dim3((DIMM * DIMM) / 256), 256, 0, stream>>>(Wqk + o1, wqkB + o1, DIMM, DIMM);
        cast_transpose_kernel<<<dim3((DIMM * DIMM) / 256), 256, 0, stream>>>(Wv  + o1, wvB  + o1, DIMM, DIMM);
        cast_transpose_kernel<<<dim3((DIMM * DIMM) / 256), 256, 0, stream>>>(Wo  + o1, woB  + o1, DIMM, DIMM);
        cast_transpose_kernel<<<dim3((DIMM * FFDIM) / 256), 256, 0, stream>>>(W1 + o2, w1B + o2, DIMM, FFDIM);
        cast_transpose_kernel<<<dim3((FFDIM * DIMM) / 256), 256, 0, stream>>>(W2 + o2, w2B + o2, FFDIM, DIMM);
    }

    init_streams_kernel<<<dim3((BT * DIMM) / 256), 256, 0, stream>>>(x, x1, x2);

    for (int l = 0; l < DEPTH; ++l) {
        size_t o1 = (size_t)l * DIMM * DIMM;
        size_t o2 = (size_t)l * DIMM * FFDIM;
        // h = LN(x2)
        layernorm_kernel<<<dim3(BT), 256, 0, stream>>>(x2, ln1g + l * DIMM, ln1b + l * DIMM, hbf);
        // qk, v projections (head-split layout [bh,T,DH])
        gemm_bf16_kernel<0, true><<<dim3(DIMM / 64, BT / 128), 256, 0, stream>>>(
            hbf, wqkB + o1, qkb, nullptr, nullptr, nullptr, BT, DIMM, DIMM);
        gemm_bf16_kernel<0, true><<<dim3(DIMM / 64, BT / 128), 256, 0, stream>>>(
            hbf, wvB + o1, vb, nullptr, nullptr, nullptr, BT, DIMM, DIMM);
        // LSH buckets (f32 exact) + stable sort
        rotate_argmax_kernel<<<dim3(SEQT, BHN), 128, 0, stream>>>(
            qkb, rot + (size_t)l * DHEAD * NHASH * 32, bkb);
        bucket_sort_kernel<<<dim3(NHASH, BHN), 256, 0, stream>>>(bkb, stb, unb);
        // chunked attention (WMMA)
        attn_chunk_kernel<<<dim3(NCHUNK, BHN), 256, 0, stream>>>(qkb, vb, stb, boff, lseb);
        // unsort + combine hash rounds -> attn activation (reuse hbf)
        combine_kernel<<<dim3(SEQT, BHN), 128, 0, stream>>>(boff, lseb, unb, hbf);
        // y1 = attn @ Wo + bo + x1
        gemm_bf16_kernel<1, false><<<dim3(DIMM / 64, BT / 128), 256, 0, stream>>>(
            hbf, woB + o1, x1, nullptr, bo + l * DIMM, x1, BT, DIMM, DIMM);
        // h2 = LN(y1)
        layernorm_kernel<<<dim3(BT), 256, 0, stream>>>(x1, ln2g + l * DIMM, ln2b + l * DIMM, hbf);
        // ff1 = gelu(h2 @ W1 + b1)  (bf16, reuses bo_sorted region)
        gemm_bf16_kernel<2, false><<<dim3(FFDIM / 64, BT / 128), 256, 0, stream>>>(
            hbf, w1B + o2, nullptr, boff, b1 + l * FFDIM, nullptr, BT, FFDIM, DIMM);
        // y2 = ff1 @ W2 + b2 + x2
        gemm_bf16_kernel<1, false><<<dim3(DIMM / 64, BT / 128), 256, 0, stream>>>(
            boff, w2B + o2, x2, nullptr, b2 + l * DIMM, x2, BT, DIMM, FFDIM);
    }

    final_kernel<<<dim3((BT * DIMM) / 256), 256, 0, stream>>>(x1, x2, (float*)d_out);
}
